// Network_41918880809426
// MI455X (gfx1250) — compile-verified, matrix operand order checked by reading, not run on hardware
//
#include <hip/hip_runtime.h>

typedef _Float16 half_t;
typedef __attribute__((ext_vector_type(16))) _Float16 v16h;
typedef __attribute__((ext_vector_type(8)))  _Float16 v8h;
typedef __attribute__((ext_vector_type(8)))  float    v8f;

#define LAYERS 24
#define BN     16384
#define LB     (LAYERS * BN)
#define D_OPD  32
#define D_ATTR 64
#define D_FILT 70
#define D_OUTD 56
#define HDIM   20
#define G4     80     // 4*H gate width
#define KX     160    // padded x inner dim (136 -> 5 chunks of 32)
#define KU     128    // padded u inner dim: [filter(0..69) | 0,0 | output(72..127)]
#define NE     48     // padded embedding width (40 -> 3 tiles of 16)
#define MT     32     // M rows per wave (2 x 16-row subtiles)

union V16U { v16h v; struct { v8h lo, hi; } s; };

// B-fragment (K=32 x N=16, f16): lane l holds column n = nbase + (l&15),
// K = k0 + (l>>4)*8 + {0..7} (lo) and +16 (hi). W row-major [N][rowstride].
__device__ __forceinline__ v16h load_bfrag(const half_t* __restrict__ W, int rowstride,
                                           int nbase, int k0, int lane) {
  int n  = nbase + (lane & 15);
  int kb = k0 + ((lane >> 4) << 3);
  V16U u;
  u.s.lo = *(const v8h*)(W + n * rowstride + kb);
  u.s.hi = *(const v8h*)(W + n * rowstride + kb + 16);
  return u.v;
}

// convert 8 contiguous f32 into A-frag slots [dst..dst+7]
__device__ __forceinline__ void cvt8(v16h& a, int dst, const float* __restrict__ p) {
  #pragma unroll
  for (int j = 0; j < 8; ++j) a[dst + j] = (half_t)p[j];
}

// one 8-run of the permuted u vector: K<70 = filter[K], K 70/71 = 0, K>=72 = output[K-72]
__device__ __forceinline__ void u_run(v16h& a, int dst, const float* __restrict__ frow,
                                      const float* __restrict__ orow, int s) {
  if (s + 8 <= D_FILT) {
    cvt8(a, dst, frow + s);
  } else if (s >= 72) {
    cvt8(a, dst, orow + (s - 72));
  } else { // s == 64: filter tail + 2 zeros
    #pragma unroll
    for (int j = 0; j < 8; ++j) a[dst + j] = (half_t)((j < 6) ? frow[64 + j] : 0.0f);
  }
}

// One wave per 32-row tile:
//   E = relu(u @ U^T + cb)            (WMMA, K=128, N=48, 2 subtiles)
//   Gx = [op|attr|E] @ W_ih^T          (WMMA, K=160, N=80, 2 subtiles)
// E routed C-layout -> A-layout via LDS.
__global__ void __launch_bounds__(256) fused_embed_gemm(
    const float* __restrict__ op, const float* __restrict__ attr,
    const float* __restrict__ filt, const float* __restrict__ outp,
    const half_t* __restrict__ Uw, const half_t* __restrict__ Ww,
    const float* __restrict__ cb, float* __restrict__ Gx) {
  __shared__ half_t ldsE[8][MT][64];
  const int lane = threadIdx.x & 31;
  const int wave = threadIdx.x >> 5;
  const int r0   = (blockIdx.x * 8 + wave) * MT;
  const int ln   = lane & 15;
  const int lh   = lane >> 4;
  const int koff = lh * 8;

  // per-subtile row base pointers
  const float* frow[2]; const float* orow[2];
  const float* oprow[2]; const float* atrow[2];
  #pragma unroll
  for (int m = 0; m < 2; ++m) {
    int row = r0 + m * 16 + ln;
    frow[m]  = filt + (size_t)row * D_FILT;
    orow[m]  = outp + (size_t)row * D_OUTD;
    oprow[m] = op   + (size_t)row * D_OPD;
    atrow[m] = attr + (size_t)row * D_ATTR;
  }
  if (r0 + MT < LB) {
    __builtin_prefetch(filt + (size_t)(r0 + MT * 8) * D_FILT, 0, 1);
    __builtin_prefetch(attr + (size_t)(r0 + MT * 8) * D_ATTR, 0, 1);
  }

  // ---- stage 1: embedding GEMM ----
  v8f e[2][3] = {{{}, {}, {}}, {{}, {}, {}}};
  #pragma unroll
  for (int kc = 0; kc < 4; ++kc) {
    int kb = kc * 32;
    v16h a[2];
    #pragma unroll
    for (int m = 0; m < 2; ++m) {
      u_run(a[m], 0, frow[m], orow[m], kb + koff);
      u_run(a[m], 8, frow[m], orow[m], kb + koff + 16);
    }
    #pragma unroll
    for (int t = 0; t < 3; ++t) {
      v16h bfr = load_bfrag(Uw, KU, t * 16, kb, lane);
      e[0][t] = __builtin_amdgcn_wmma_f32_16x16x32_f16(
                    false, a[0], false, bfr, (short)0, e[0][t], false, false);
      e[1][t] = __builtin_amdgcn_wmma_f32_16x16x32_f16(
                    false, a[1], false, bfr, (short)0, e[1][t], false, false);
    }
  }

  // relu + bias, f16 into LDS row-major [MT][64] (cols 40..63 zero)
  #pragma unroll
  for (int m = 0; m < 2; ++m) {
    #pragma unroll
    for (int t = 0; t < 3; ++t) {
      #pragma unroll
      for (int v = 0; v < 8; ++v) {
        int row = m * 16 + v + lh * 8;
        int col = t * 16 + ln;
        float val = e[m][t][v] + cb[col];
        val = fmaxf(val, 0.0f);
        if (t == 2 && ln >= 8) val = 0.0f;       // cols 40..47 -> pad
        ldsE[wave][row][col] = (half_t)val;
      }
    }
    v8h z = {};
    *(v8h*)&ldsE[wave][m * 16 + ln][48 + lh * 8] = z; // cols 48..63 -> pad
  }
  __syncthreads();

  // ---- stage 2: gate GEMM, x = [op(0..31)|attr(32..95)|E(96..159)] ----
  v8f g[2][5] = {{{}, {}, {}, {}, {}}, {{}, {}, {}, {}, {}}};
  #pragma unroll
  for (int kc = 0; kc < 5; ++kc) {
    v16h a[2];
    #pragma unroll
    for (int m = 0; m < 2; ++m) {
      if (kc == 0) {
        cvt8(a[m], 0, oprow[m] + koff);
        cvt8(a[m], 8, oprow[m] + koff + 16);
      } else if (kc <= 2) {
        int kb = (kc - 1) * 32;
        cvt8(a[m], 0, atrow[m] + kb + koff);
        cvt8(a[m], 8, atrow[m] + kb + koff + 16);
      } else {
        V16U u;
        const half_t* p = &ldsE[wave][m * 16 + ln][(kc - 3) * 32 + koff];
        u.s.lo = *(const v8h*)p;
        u.s.hi = *(const v8h*)(p + 16);
        a[m] = u.v;
      }
    }
    #pragma unroll
    for (int t = 0; t < 5; ++t) {
      v16h bfr = load_bfrag(Ww, KX, t * 16, kc * 32, lane);
      g[0][t] = __builtin_amdgcn_wmma_f32_16x16x32_f16(
                    false, a[0], false, bfr, (short)0, g[0][t], false, false);
      g[1][t] = __builtin_amdgcn_wmma_f32_16x16x32_f16(
                    false, a[1], false, bfr, (short)0, g[1][t], false, false);
    }
  }

  #pragma unroll
  for (int m = 0; m < 2; ++m)
    #pragma unroll
    for (int t = 0; t < 5; ++t)
      #pragma unroll
      for (int v = 0; v < 8; ++v)
        Gx[(size_t)(r0 + m * 16 + v + lh * 8) * G4 + t * 16 + ln] = g[m][t][v];
}

// Build padded f16 B-matrices (with the permuted u-K layout) + combined biases.
__global__ void prep_weights(const float* __restrict__ Wf, const float* __restrict__ bf,
                             const float* __restrict__ Wo, const float* __restrict__ bo,
                             const float* __restrict__ Wih, const float* __restrict__ bih,
                             const float* __restrict__ bhh,
                             half_t* __restrict__ Uw, half_t* __restrict__ Ww,
                             float* __restrict__ cb, float* __restrict__ gb) {
  int tid = threadIdx.x;
  for (int i = tid; i < NE * KU; i += 256) {
    int n = i / KU, k = i % KU;
    float v = 0.0f;
    if (n < HDIM) { if (k < D_FILT) v = Wf[n * D_FILT + k]; }
    else if (n < 2 * HDIM) {
      if (k >= 72) v = Wo[(n - HDIM) * D_OUTD + (k - 72)];
    }
    Uw[i] = (half_t)v;
  }
  for (int i = tid; i < G4 * KX; i += 256) {
    int n = i / KX, k = i % KX;
    Ww[i] = (half_t)(k < 136 ? Wih[n * 136 + k] : 0.0f);
  }
  for (int i = tid; i < NE; i += 256)
    cb[i] = (i < HDIM) ? bf[i] : (i < 2 * HDIM ? bo[i - HDIM] : 0.0f);
  for (int i = tid; i < G4; i += 256)
    gb[i] = bih[i] + bhh[i];
}

__global__ void zero_f32(float* __restrict__ p, int n) {
  int i = blockIdx.x * blockDim.x + threadIdx.x;
  int stride = gridDim.x * blockDim.x;
  for (; i < n; i += stride) p[i] = 0.0f;
}

__device__ __forceinline__ float sigf(float x) { return 1.0f / (1.0f + __expf(-x)); }
__device__ __forceinline__ float tanhfast(float x) { return 2.0f / (1.0f + __expf(-2.0f * x)) - 1.0f; }

// One thread per node: child gather, recurrent 20->80 GEMV from LDS, LSTM cell.
__global__ void __launch_bounds__(256) lstm_step(
    const float* __restrict__ Gx, const int* __restrict__ map,
    const float* __restrict__ Whh, const float* __restrict__ gb,
    const float* __restrict__ h_prev, const float* __restrict__ c_prev,
    float* __restrict__ h_out, float* __restrict__ c_out, int layer) {
  __shared__ float sW[G4 * HDIM];
  __shared__ float sB[G4];
  for (int i = threadIdx.x; i < G4 * HDIM; i += 256) sW[i] = Whh[i];
  for (int i = threadIdx.x; i < G4; i += 256) sB[i] = gb[i];
  __syncthreads();

  int b = blockIdx.x * 256 + threadIdx.x;
  const int* mp = map + (size_t)layer * BN * 2 + (size_t)b * 2;
  int m0 = mp[0], m1 = mp[1];
  float w0 = m0 != 0 ? 0.5f : 0.0f;
  float w1 = m1 != 0 ? 0.5f : 0.0f;
  int i0 = (m0 > 0 ? m0 - 1 : 0) * HDIM;
  int i1 = (m1 > 0 ? m1 - 1 : 0) * HDIM;

  float h0[HDIM], c0[HDIM];
  #pragma unroll
  for (int j = 0; j < HDIM; ++j) {
    h0[j] = w0 * h_prev[i0 + j] + w1 * h_prev[i1 + j];
    c0[j] = w0 * c_prev[i0 + j] + w1 * c_prev[i1 + j];
  }

  const float* gx = Gx + (size_t)layer * BN * G4 + (size_t)b * G4;
  #pragma unroll 4
  for (int j = 0; j < HDIM; ++j) {
    float gi = gx[j]      + sB[j];
    float gf = gx[20 + j] + sB[20 + j];
    float gg = gx[40 + j] + sB[40 + j];
    float go = gx[60 + j] + sB[60 + j];
    #pragma unroll
    for (int k = 0; k < HDIM; ++k) {
      float hv = h0[k];
      gi += hv * sW[j * HDIM + k];
      gf += hv * sW[(20 + j) * HDIM + k];
      gg += hv * sW[(40 + j) * HDIM + k];
      go += hv * sW[(60 + j) * HDIM + k];
    }
    float cc = sigf(gf) * c0[j] + sigf(gi) * tanhfast(gg);
    h_out[(size_t)b * HDIM + j] = sigf(go) * tanhfast(cc);
    c_out[(size_t)b * HDIM + j] = cc;
  }
}

__global__ void final_proj(const float* __restrict__ h, const float* __restrict__ Wfin,
                           const float* __restrict__ bfin, float* __restrict__ out) {
  int b = blockIdx.x * 256 + threadIdx.x;
  float acc = bfin[0];
  #pragma unroll
  for (int j = 0; j < HDIM; ++j) acc += h[(size_t)b * HDIM + j] * Wfin[j];
  out[b] = acc;
}

extern "C" void kernel_launch(void* const* d_in, const int* in_sizes, int n_in,
                              void* d_out, int out_size, void* d_ws, size_t ws_size,
                              hipStream_t stream) {
  (void)in_sizes; (void)n_in; (void)out_size; (void)ws_size;
  const float* op   = (const float*)d_in[0];
  const float* attr = (const float*)d_in[1];
  const float* filt = (const float*)d_in[2];
  const float* outp = (const float*)d_in[3];
  const int*   mp   = (const int*)d_in[4];
  const float* Wf   = (const float*)d_in[5];
  const float* bf   = (const float*)d_in[6];
  const float* Wo   = (const float*)d_in[7];
  const float* bo   = (const float*)d_in[8];
  const float* Wih  = (const float*)d_in[9];
  const float* Whh  = (const float*)d_in[10];
  const float* bih  = (const float*)d_in[11];
  const float* bhh  = (const float*)d_in[12];
  const float* Wfin = (const float*)d_in[13];
  const float* bfin = (const float*)d_in[14];

  char* ws = (char*)d_ws;
  size_t off = 0;
  float*  Gx = (float*)(ws + off);  off += (size_t)LB * G4 * 4;   // 125.8 MB
  half_t* Uw = (half_t*)(ws + off); off += (size_t)NE * KU * 2;   // 12 KB
  half_t* Ww = (half_t*)(ws + off); off += (size_t)G4 * KX * 2;   // 25.6 KB
  float*  cb = (float*)(ws + off);  off += 256;
  float*  gb = (float*)(ws + off);  off += 512;
  float* hbuf[2]; float* cbuf[2];
  hbuf[0] = (float*)(ws + off); off += (size_t)BN * HDIM * 4;
  cbuf[0] = (float*)(ws + off); off += (size_t)BN * HDIM * 4;
  hbuf[1] = (float*)(ws + off); off += (size_t)BN * HDIM * 4;
  cbuf[1] = (float*)(ws + off); off += (size_t)BN * HDIM * 4;

  prep_weights<<<1, 256, 0, stream>>>(Wf, bf, Wo, bo, Wih, bih, bhh, Uw, Ww, cb, gb);
  zero_f32<<<512, 256, 0, stream>>>(hbuf[0], BN * HDIM);
  zero_f32<<<512, 256, 0, stream>>>(cbuf[0], BN * HDIM);

  fused_embed_gemm<<<LB / (8 * MT), 256, 0, stream>>>(op, attr, filt, outp, Uw, Ww, cb, Gx);

  int cur = 0;
  for (int l = LAYERS - 1; l >= 0; --l) {
    int nxt = cur ^ 1;
    lstm_step<<<BN / 256, 256, 0, stream>>>(Gx, mp, Whh, gb,
                                            hbuf[cur], cbuf[cur],
                                            hbuf[nxt], cbuf[nxt], l);
    cur = nxt;
  }
  final_proj<<<BN / 256, 256, 0, stream>>>(hbuf[cur], Wfin, bfin, (float*)d_out);
}
